// CustomLSTM_1348619731408
// MI455X (gfx1250) — matrix operand order repaired and checked by reading, not compiled, not run on hardware
//
#include <hip/hip_runtime.h>

typedef __bf16 bf16_t;
typedef __attribute__((ext_vector_type(16))) __bf16 v16bf;
typedef __attribute__((ext_vector_type(8)))  __bf16 v8bf;
typedef __attribute__((ext_vector_type(4)))  __bf16 v4bf;
typedef __attribute__((ext_vector_type(8)))  float  v8f;

#define BATCH 64
#define SEQ   512
#define ISZ   1024
#define HSZ   1024
#define K2    2048   // concat [x_t ; h_prev]
#define NSTEP (K2 / 32)   // 64 K-steps of 32

// d_out layout (flat, return order): h_t, hidden_seq, c_t, carry_seq
#define OFF_HT     ((size_t)0)
#define OFF_HIDDEN ((size_t)BATCH * HSZ)                               // 65536
#define OFF_CT     (OFF_HIDDEN + (size_t)BATCH * SEQ * HSZ)            // 33619968
#define OFF_CARRY  (OFF_CT + (size_t)BATCH * HSZ)                      // 33685504

#if __has_builtin(__builtin_amdgcn_sched_barrier)
#define SCHED_FENCE() __builtin_amdgcn_sched_barrier(0)
#else
#define SCHED_FENCE()
#endif

union AFrag { v16bf v; v8bf h[2]; };

__device__ __forceinline__ float fast_sigmoid(float x) {
    return 1.0f / (1.0f + __expf(-x));
}
__device__ __forceinline__ float fast_tanh(float x) {
    x = fminf(15.0f, fmaxf(-15.0f, x));
    float e = __expf(2.0f * x);
    return (e - 1.0f) / (e + 1.0f);
}

#define WMMA_BF16(A, Bv, C) \
    __builtin_amdgcn_wmma_f32_16x16x32_bf16(false, (A), false, (Bv), (short)0, (C), false, false)

// A fragment (16-bit A 16x32): elem j -> K=k0+half*16+j (matches B)
__device__ __forceinline__ AFrag load_afrag(const bf16_t* __restrict__ row, int half) {
    AFrag a;
    a.h[0] = *(const v8bf*)(row + half * 16);
    a.h[1] = *(const v8bf*)(row + half * 16 + 8);
    return a;
}
// B fragment (16-bit B 32x16, contiguous k order as in A): elem j -> K=k0+half*16+j, N=lane15 (half*16 folded into base)
__device__ __forceinline__ v16bf load_bfrag(const bf16_t* __restrict__ p) {
    return *(const v16bf*)p;
}

// Build transposed bf16 weights, gate-interleaved for small immediate offsets:
//   UWT[h][gate][k] ; this kernel writes one gate: dst = UWT + gate*K2,
//   dst[h*(4*K2) + k] = (k < 1024 ? U[k][h] : W[k-1024][h])
__global__ void prep_gate_kernel(const float* __restrict__ U,
                                 const float* __restrict__ W,
                                 bf16_t* __restrict__ dst) {
    const int total = HSZ * K2;
    for (int idx = blockIdx.x * blockDim.x + threadIdx.x; idx < total;
         idx += gridDim.x * blockDim.x) {
        int h = idx >> 11;
        int k = idx & (K2 - 1);
        float v = (k < ISZ) ? U[(size_t)k * HSZ + h] : W[(size_t)(k - ISZ) * HSZ + h];
        dst[(size_t)h * (4 * K2) + k] = (bf16_t)v;
    }
}

// Convert x (fp32) -> bf16, vectorized.
__global__ void convert_x_kernel(const float4* __restrict__ x4, v4bf* __restrict__ out4) {
    const int total4 = (BATCH * SEQ * ISZ) / 4;   // 8,388,608
    for (int i = blockIdx.x * blockDim.x + threadIdx.x; i < total4;
         i += gridDim.x * blockDim.x) {
        float4 v = x4[i];
        v4bf o = { (bf16_t)v.x, (bf16_t)v.y, (bf16_t)v.z, (bf16_t)v.w };
        out4[i] = o;
    }
}

// Zero the bf16 h ping-pong buffers (2 * 64 * 1024 bf16 = 65536 u32 words).
__global__ void zero_h_kernel(unsigned int* __restrict__ p) {
    const int total = (2 * BATCH * HSZ * 2) / 4;
    for (int i = blockIdx.x * blockDim.x + threadIdx.x; i < total;
         i += gridDim.x * blockDim.x)
        p[i] = 0u;
}

// One LSTM timestep. grid = 64 blocks x 128 threads = 256 waves.
// Each wave computes ONE 16x16 output tile for ALL FOUR gates (f,i,o,g).
// Fully unrolled K-loop, software-pipelined with prefetch distance 2 and
// sched_barriers so ~20 loads stay in flight ahead of each WMMA group.
// LSTM elementwise combine runs entirely in registers (no LDS, no barrier).
__global__ void __launch_bounds__(128)
lstm_step_kernel(const bf16_t* __restrict__ xbf,     // [B][S][1024] bf16
                 const bf16_t* __restrict__ UWT,     // [1024 h][4 gate][2048 k] bf16
                 const bf16_t* __restrict__ hprev,   // [B][1024] bf16 (read)
                 bf16_t* __restrict__ hnext,         // [B][1024] bf16 (write)
                 const float* __restrict__ bf_, const float* __restrict__ bi_,
                 const float* __restrict__ bo_, const float* __restrict__ bg_,
                 float* __restrict__ out, int t) {
    const int lane   = threadIdx.x & 31;
    const int wv     = threadIdx.x >> 5;
    const int tile   = blockIdx.x * 4 + wv;   // 0..255
    const int mt     = tile & 3;              // batch rows [mt*16, +16)
    const int ncol   = tile >> 2;             // h columns [ncol*16, +16)
    const int lane15 = lane & 15;
    const int half   = lane >> 4;
    const int brow   = mt * 16 + lane15;
    const int hcol   = ncol * 16 + lane15;    // this lane's output column

    const bf16_t* arow_x = xbf + (size_t)brow * SEQ * ISZ + (size_t)t * ISZ;
    const bf16_t* arow_h = hprev + (size_t)brow * HSZ;
    // Gate g's weight row lives at bk + g*K2 (4 KB apart -> immediate offsets off one base).
    const bf16_t* bk = UWT + (size_t)hcol * (4 * K2) + half * 16;

    v8f accf = {}, acci = {}, acco = {}, accg = {};

    // Two pipeline slots, prefetch distance 2. Step s covers K = [s*32, s*32+32);
    // steps 0..31 read A from x, steps 32..63 from h_prev (folds at compile time).
#define AROW(s) (((s) * 32 < ISZ) ? (arow_x + (s) * 32) : (arow_h + ((s) * 32 - ISZ)))
    AFrag a0 = load_afrag(AROW(0), half);
    v16bf p00 = load_bfrag(bk + 0 * K2);
    v16bf p01 = load_bfrag(bk + 1 * K2);
    v16bf p02 = load_bfrag(bk + 2 * K2);
    v16bf p03 = load_bfrag(bk + 3 * K2);
    AFrag a1 = load_afrag(AROW(1), half);
    v16bf p10 = load_bfrag(bk + 0 * K2 + 32);
    v16bf p11 = load_bfrag(bk + 1 * K2 + 32);
    v16bf p12 = load_bfrag(bk + 2 * K2 + 32);
    v16bf p13 = load_bfrag(bk + 3 * K2 + 32);

#pragma unroll
    for (int s = 0; s < NSTEP; ++s) {
        // consume slot 0 (operands for step s)
        accf = WMMA_BF16(a0.v, p00, accf);
        acci = WMMA_BF16(a0.v, p01, acci);
        acco = WMMA_BF16(a0.v, p02, acco);
        accg = WMMA_BF16(a0.v, p03, accg);
        if (s + 2 < NSTEP) {                 // refill freed slot with step s+2
            int kn = (s + 2) * 32;
            a0  = load_afrag(AROW(s + 2), half);
            p00 = load_bfrag(bk + 0 * K2 + kn);
            p01 = load_bfrag(bk + 1 * K2 + kn);
            p02 = load_bfrag(bk + 2 * K2 + kn);
            p03 = load_bfrag(bk + 3 * K2 + kn);
        }
        SCHED_FENCE();                       // keep these loads ahead of next WMMA group
        // rotate slots (pure renaming after full unroll)
        AFrag ta = a0; a0 = a1; a1 = ta;
        v16bf tb;
        tb = p00; p00 = p10; p10 = tb;
        tb = p01; p01 = p11; p11 = tb;
        tb = p02; p02 = p12; p12 = tb;
        tb = p03; p03 = p13; p13 = tb;
    }
#undef AROW

    // Per-lane biases: every element this lane owns has column hcol.
    const float biasf = bf_[hcol];
    const float biasi = bi_[hcol];
    const float biaso = bo_[hcol];
    const float biasg = bg_[hcol];

    // Elementwise LSTM combine in registers.
    // C/D layout: acc[v] is (row = v + half*8, col = lane15) of the 16x16 tile.
    float* hidden = out + OFF_HIDDEN;
    float* carry  = out + OFF_CARRY;
    float* h_t    = out + OFF_HT;
    float* c_t    = out + OFF_CT;
#pragma unroll
    for (int v = 0; v < 8; ++v) {
        int b = mt * 16 + v + half * 8;
        float fv = fast_sigmoid(accf[v] + biasf);
        float iv = fast_sigmoid(acci[v] + biasi);
        float ov = fast_sigmoid(acco[v] + biaso);
        float gv = fast_tanh   (accg[v] + biasg);
        size_t base  = (size_t)b * SEQ * HSZ + (size_t)t * HSZ + hcol;
        float cprev  = (t == 0) ? 0.0f : carry[base - HSZ];
        float cv     = fv * cprev + iv * gv;
        float hv     = ov * fast_tanh(cv);
        hidden[base] = hv;
        carry[base]  = cv;
        hnext[(size_t)b * HSZ + hcol] = (bf16_t)hv;
        if (t == SEQ - 1) {
            h_t[(size_t)b * HSZ + hcol] = hv;
            c_t[(size_t)b * HSZ + hcol] = cv;
        }
    }
}

extern "C" void kernel_launch(void* const* d_in, const int* in_sizes, int n_in,
                              void* d_out, int out_size, void* d_ws, size_t ws_size,
                              hipStream_t stream) {
    const float* x = (const float*)d_in[0];
    // dict order: x, then (U,W,b) for f,i,o,g
    const float* U[4] = { (const float*)d_in[1], (const float*)d_in[4],
                          (const float*)d_in[7], (const float*)d_in[10] };
    const float* W[4] = { (const float*)d_in[2], (const float*)d_in[5],
                          (const float*)d_in[8], (const float*)d_in[11] };
    const float* B[4] = { (const float*)d_in[3], (const float*)d_in[6],
                          (const float*)d_in[9], (const float*)d_in[12] };
    float* out = (float*)d_out;

    // Workspace layout (bytes):
    //   [0, 16 MB)        UWT  : [1024 h][4 gate][2048 k] bf16
    //   [16 MB, 80 MB)    xbf  : 64*512*1024 bf16
    //   [80 MB, +256 KB)  hbuf : 2 * 64*1024 bf16 ping-pong
    char* ws = (char*)d_ws;
    const size_t UWT_BYTES = (size_t)HSZ * 4 * K2 * sizeof(bf16_t);
    const size_t XBF_BYTES = (size_t)BATCH * SEQ * ISZ * sizeof(bf16_t);
    bf16_t* UWT  = (bf16_t*)ws;
    bf16_t* xbf  = (bf16_t*)(ws + UWT_BYTES);
    bf16_t* hbuf = (bf16_t*)(ws + UWT_BYTES + XBF_BYTES);

    for (int g = 0; g < 4; ++g)
        prep_gate_kernel<<<2048, 256, 0, stream>>>(U[g], W[g], UWT + (size_t)g * K2);
    convert_x_kernel<<<4096, 256, 0, stream>>>((const float4*)x, (v4bf*)xbf);
    zero_h_kernel<<<64, 256, 0, stream>>>((unsigned int*)hbuf);

    const size_t HB = (size_t)BATCH * HSZ;   // elems per h buffer
    for (int t = 0; t < SEQ; ++t) {
        lstm_step_kernel<<<64, 128, 0, stream>>>(
            xbf, UWT,
            hbuf + (size_t)(t & 1) * HB,         // read h_{t-1}
            hbuf + (size_t)((t + 1) & 1) * HB,   // write h_t
            B[0], B[1], B[2], B[3],
            out, t);
    }
}
